// DCGN_64424509440377
// MI455X (gfx1250) — matrix operand order repaired and checked by reading, not compiled, not run on hardware
//
#include <hip/hip_runtime.h>
#include <hip/hip_bf16.h>

typedef __attribute__((ext_vector_type(16))) __bf16 v16bf;
typedef __attribute__((ext_vector_type(8)))  float  v8f;

__device__ __forceinline__ __bf16 to_bf16(float f) { return (__bf16)f; }

// ---- CDNA5 async global->LDS copy (16B per lane), tracked by ASYNCcnt ------
__device__ __forceinline__ void async_copy_b128(void* lds_dst, const void* gsrc)
{
  const unsigned l = (unsigned)(unsigned long long)lds_dst;  // low 32 bits = LDS offset
  asm volatile("global_load_async_to_lds_b128 %0, %1, off"
               :: "v"(l), "v"(gsrc)
               : "memory");
}
__device__ __forceinline__ void wait_async0()
{
  asm volatile("s_wait_asynccnt 0" ::: "memory");
}

// ---------------------------------------------------------------------------
// Attention pooling over non-overlapping windows of 3 rows.
// ---------------------------------------------------------------------------
__global__ __launch_bounds__(256) void attpool_kernel(
    const float* __restrict__ x, const float* __restrict__ w,
    const float* __restrict__ b, float* __restrict__ out, int F)
{
  const int s   = blockIdx.x;
  const int tid = threadIdx.x;
  const float* xr = x + (size_t)s * 3 * F;

  __shared__ float red[3][256];
  float l0 = 0.f, l1 = 0.f, l2 = 0.f;
  for (int f = tid; f < F; f += 256) {
    const float wf = w[f];
    l0 += xr[f] * wf;
    l1 += xr[F + f] * wf;
    l2 += xr[2 * F + f] * wf;
  }
  red[0][tid] = l0; red[1][tid] = l1; red[2][tid] = l2;
  __syncthreads();
  for (int st = 128; st > 0; st >>= 1) {
    if (tid < st) {
      red[0][tid] += red[0][tid + st];
      red[1][tid] += red[1][tid + st];
      red[2][tid] += red[2][tid + st];
    }
    __syncthreads();
  }
  __shared__ float att[3];
  if (tid == 0) {
    const float bb = b[0];
    const float a0 = red[0][0] + bb, a1 = red[1][0] + bb, a2 = red[2][0] + bb;
    const float m  = fmaxf(a0, fmaxf(a1, a2));
    const float e0 = __expf(a0 - m), e1 = __expf(a1 - m), e2 = __expf(a2 - m);
    const float inv = 1.f / (e0 + e1 + e2);
    att[0] = e0 * inv; att[1] = e1 * inv; att[2] = e2 * inv;
  }
  __syncthreads();
  const float a0 = att[0], a1 = att[1], a2 = att[2];
  for (int f = tid; f < F; f += 256)
    out[(size_t)s * F + f] = a0 * xr[f] + a1 * xr[F + f] + a2 * xr[2 * F + f];
}

// ---------------------------------------------------------------------------
// Row L2-normalize + bf16 cast:  q = pooled / |pooled|
// ---------------------------------------------------------------------------
__global__ __launch_bounds__(256) void rownorm_kernel(
    const float* __restrict__ in, __bf16* __restrict__ out, int F)
{
  const int s = blockIdx.x, tid = threadIdx.x;
  const float* r = in + (size_t)s * F;
  __shared__ float red[256];
  float acc = 0.f;
  for (int f = tid; f < F; f += 256) { const float v = r[f]; acc += v * v; }
  red[tid] = acc;
  __syncthreads();
  for (int st = 128; st > 0; st >>= 1) {
    if (tid < st) red[tid] += red[tid + st];
    __syncthreads();
  }
  __shared__ float invn;
  if (tid == 0) invn = rsqrtf(red[0]);
  __syncthreads();
  const float sc = invn;
  for (int f = tid; f < F; f += 256)
    out[(size_t)s * F + f] = to_bf16(r[f] * sc);
}

// ---------------------------------------------------------------------------
// dinv[i] = 1/sqrt(sum_j A[i,j])   (A symmetric -> row sums == col sums)
// ---------------------------------------------------------------------------
__global__ __launch_bounds__(256) void rowsum_dinv_kernel(
    const __bf16* __restrict__ A, float* __restrict__ dinv, int S)
{
  const int i = blockIdx.x, tid = threadIdx.x;
  const __bf16* r = A + (size_t)i * S;
  __shared__ float red[256];
  float acc = 0.f;
  for (int j = tid; j < S; j += 256) acc += (float)r[j];
  red[tid] = acc;
  __syncthreads();
  for (int st = 128; st > 0; st >>= 1) {
    if (tid < st) red[tid] += red[tid + st];
    __syncthreads();
  }
  if (tid == 0) dinv[i] = rsqrtf(red[0]);
}

// ---------------------------------------------------------------------------
// Node conv fused with dinv row-scale, TRANSPOSED bf16 output (K-contiguous
// for the NT GEMM):  outT[f,s] = bf16( dinv[s] * sum_p x[3s+p,f] * w[p,f] )
// Coalesced on the store side.
// ---------------------------------------------------------------------------
__global__ __launch_bounds__(256) void nodeconv_t_kernel(
    const float* __restrict__ x, const float* __restrict__ w,
    const float* __restrict__ dinv, __bf16* __restrict__ outT, int S, int F)
{
  const int idx = blockIdx.x * 256 + threadIdx.x;
  if (idx >= S * F) return;
  const int f = idx / S, s = idx % S;
  const float* xr = x + (size_t)s * 3 * F + f;
  const float v = xr[0] * w[f] + xr[F] * w[F + f] + xr[2 * F] * w[2 * F + f];
  outT[(size_t)f * S + s] = to_bf16(v * dinv[s]);
}

// ---------------------------------------------------------------------------
// f32 [K,N] -> bf16 transposed [N,K]
// ---------------------------------------------------------------------------
__global__ __launch_bounds__(256) void f2bf_t_kernel(
    const float* __restrict__ in, __bf16* __restrict__ outT, int K, int N)
{
  const int idx = blockIdx.x * 256 + threadIdx.x;
  if (idx >= K * N) return;
  const int n = idx / K, k = idx % K;
  outT[(size_t)n * K + k] = to_bf16(in[(size_t)k * N + n]);
}

// ---------------------------------------------------------------------------
// NT bf16 GEMM, f32 WMMA accumulation:  C[M,N] = A[M,K] * B[N,K]^T
// Block tile 128x128, K-step 32, double-buffered LDS filled by
// global_load_async_to_lds_b128 (ASYNCcnt), 8 waves each computing 32x64
// (2 A-frags x 4 B-frags = 8 v_wmma_f32_16x16x32_bf16 per K-step).
// EPI: 1 = bf16 out + identity, 2 = bf16 out * rowscale[row], 3 = f32 ELU.
// Requires M,N % 128 == 0, K % 32 == 0 (true for all calls here).
// ---------------------------------------------------------------------------
template <int EPI>
__global__ __launch_bounds__(256) void gemm_nt_bf16_kernel(
    const __bf16* __restrict__ A, const __bf16* __restrict__ B,
    void* __restrict__ Cv, const float* __restrict__ rowscale,
    int M, int N, int K, int lda, int ldb, int ldc)
{
  __shared__ alignas(16) __bf16 As[2][128][32];
  __shared__ alignas(16) __bf16 Bs[2][128][32];

  const int tid  = threadIdx.x;
  const int m0   = blockIdx.y * 128;
  const int n0   = blockIdx.x * 128;
  const int lane = tid & 31;
  const int wid  = tid >> 5;
  const int wm   = (wid & 3) << 5;    // 0,32,64,96
  const int wn   = (wid >> 2) << 6;   // 0,64
  const int half = lane >> 4;
  const int l16  = lane & 15;

  // staging map: each thread async-copies 2x16B per tile
  const int sm = tid >> 1;            // 0..127
  const int sk = (tid & 1) << 4;      // 0,16

  const v8f vz = {0.f, 0.f, 0.f, 0.f, 0.f, 0.f, 0.f, 0.f};
  v8f acc[2][4];
  #pragma unroll
  for (int a = 0; a < 2; ++a)
    #pragma unroll
    for (int c = 0; c < 4; ++c) acc[a][c] = vz;

  const __bf16* gA = A + (size_t)(m0 + sm) * lda + sk;
  const __bf16* gB = B + (size_t)(n0 + sm) * ldb + sk;

  // prologue: prefetch tile 0 into buffer 0
  async_copy_b128(&As[0][sm][sk],     gA);
  async_copy_b128(&As[0][sm][sk + 8], gA + 8);
  async_copy_b128(&Bs[0][sm][sk],     gB);
  async_copy_b128(&Bs[0][sm][sk + 8], gB + 8);

  const int nk = K >> 5;
  for (int i = 0; i < nk; ++i) {
    const int cur = i & 1;
    wait_async0();        // own prefetch of tile i landed
    __syncthreads();      // everyone's tile i landed; tile i-1 fully consumed

    if (i + 1 < nk) {     // prefetch tile i+1 into the other buffer
      const int k0 = (i + 1) << 5;
      async_copy_b128(&As[cur ^ 1][sm][sk],     gA + k0);
      async_copy_b128(&As[cur ^ 1][sm][sk + 8], gA + k0 + 8);
      async_copy_b128(&Bs[cur ^ 1][sm][sk],     gB + k0);
      async_copy_b128(&Bs[cur ^ 1][sm][sk + 8], gB + k0 + 8);
    }

    // fragment loads (VGPR layouts per CDNA5 ISA 7.12.2 / 7.12.4)
    union { v16bf v; unsigned int u[8]; } fa[2], fb[4];
    #pragma unroll
    for (int r = 0; r < 8; ++r) {
      const int kka = ((r & 3) << 1) + ((r >> 2) << 4) + (half << 3);
      const int kkb = ((r & 3) << 1) + ((r >> 2) << 4) + (half << 3);
      fa[0].u[r] = *(const unsigned int*)(&As[cur][wm + l16][kka]);
      fa[1].u[r] = *(const unsigned int*)(&As[cur][wm + 16 + l16][kka]);
      #pragma unroll
      for (int c = 0; c < 4; ++c)
        fb[c].u[r] = *(const unsigned int*)(&Bs[cur][wn + c * 16 + l16][kkb]);
    }
    #pragma unroll
    for (int a = 0; a < 2; ++a)
      #pragma unroll
      for (int c = 0; c < 4; ++c)
        acc[a][c] = __builtin_amdgcn_wmma_f32_16x16x32_bf16(
            false, fa[a].v, false, fb[c].v, (short)0, acc[a][c], false, false);
  }

  // epilogue: VGPR r -> row (r + 8*half), col = lane&15 within each 16x16 frag
  #pragma unroll
  for (int a = 0; a < 2; ++a) {
    const int row0 = m0 + wm + a * 16 + half * 8;
    #pragma unroll
    for (int c = 0; c < 4; ++c) {
      const int col = n0 + wn + c * 16 + l16;
      #pragma unroll
      for (int r = 0; r < 8; ++r) {
        const int row = row0 + r;
        float v = acc[a][c][r];
        if (EPI == 1) {
          if (row == col) v += 1.f;
          ((__bf16*)Cv)[(size_t)row * ldc + col] = to_bf16(v);
        } else if (EPI == 2) {
          ((__bf16*)Cv)[(size_t)row * ldc + col] = to_bf16(v * rowscale[row]);
        } else {
          ((float*)Cv)[(size_t)row * ldc + col] =
              v > 0.f ? v : (__expf(v) - 1.f);
        }
      }
    }
  }
}

// ---------------------------------------------------------------------------
// MoE head: top-4 gating (low-index tie-break like lax.top_k) + expert MLPs
// ---------------------------------------------------------------------------
__global__ __launch_bounds__(64) void moe_kernel(
    const float* __restrict__ x,  const float* __restrict__ wg,
    const float* __restrict__ w1, const float* __restrict__ b1,
    const float* __restrict__ w2, const float* __restrict__ b2,
    float* __restrict__ out)
{
  const int n = blockIdx.x, tid = threadIdx.x;
  __shared__ float xr[512];
  __shared__ float lg[10];
  __shared__ float h[64];
  __shared__ float ol[10];
  __shared__ float oacc[10];
  __shared__ int   sel[4];
  __shared__ float gv[4];

  for (int d = tid; d < 512; d += 64) xr[d] = x[(size_t)n * 512 + d];
  __syncthreads();

  if (tid < 10) {
    float s = 0.f;
    for (int d = 0; d < 512; ++d) s += xr[d] * wg[d * 10 + tid];
    lg[tid]   = s;
    oacc[tid] = 0.f;
  }
  __syncthreads();

  if (tid == 0) {
    bool used[10] = {};
    float tv[4];
    for (int k = 0; k < 4; ++k) {
      float best = -3.4e38f; int bi = 0;
      for (int e = 0; e < 10; ++e)
        if (!used[e] && lg[e] > best) { best = lg[e]; bi = e; }
      used[bi] = true; sel[k] = bi; tv[k] = best;
    }
    float m = tv[0];
    for (int k = 1; k < 4; ++k) m = fmaxf(m, tv[k]);
    float ssum = 0.f, ev[4];
    for (int k = 0; k < 4; ++k) { ev[k] = __expf(tv[k] - m); ssum += ev[k]; }
    for (int k = 0; k < 4; ++k) gv[k] = ev[k] / ssum;
  }
  __syncthreads();

  for (int k = 0; k < 4; ++k) {
    const int e = sel[k];
    {
      const float* w = w1 + (size_t)e * 512 * 64 + tid;
      float s = b1[e * 64 + tid];
      for (int d = 0; d < 512; ++d) s += xr[d] * w[(size_t)d * 64];
      h[tid] = fmaxf(s, 0.f);
    }
    __syncthreads();
    if (tid < 10) {
      const float* w = w2 + (size_t)e * 64 * 10 + tid;
      float s = b2[e * 10 + tid];
      for (int d = 0; d < 64; ++d) s += h[d] * w[d * 10];
      ol[tid] = s;
    }
    __syncthreads();
    if (tid == 0) {
      float m = ol[0];
      for (int o = 1; o < 10; ++o) m = fmaxf(m, ol[o]);
      float ssum = 0.f, ev[10];
      for (int o = 0; o < 10; ++o) { ev[o] = __expf(ol[o] - m); ssum += ev[o]; }
      const float g = gv[k];
      for (int o = 0; o < 10; ++o) oacc[o] += g * ev[o] / ssum;
    }
    __syncthreads();
  }
  if (tid < 10) out[(size_t)n * 10 + tid] = oacc[tid];
}

// ---------------------------------------------------------------------------
extern "C" void kernel_launch(void* const* d_in, const int* in_sizes, int n_in,
                              void* d_out, int out_size, void* d_ws, size_t ws_size,
                              hipStream_t stream)
{
  (void)in_sizes; (void)n_in; (void)out_size; (void)ws_size;

  const float* x     = (const float*)d_in[0];
  const float* wp_w  = (const float*)d_in[1];
  const float* wp_b  = (const float*)d_in[2];
  const float* nc0_w = (const float*)d_in[3];
  const float* P1    = (const float*)d_in[4];
  const float* ap1_w = (const float*)d_in[5];
  const float* ap1_b = (const float*)d_in[6];
  const float* nc1_w = (const float*)d_in[7];
  const float* P2    = (const float*)d_in[8];
  const float* ap2_w = (const float*)d_in[9];
  const float* ap2_b = (const float*)d_in[10];
  const float* nc2_w = (const float*)d_in[11];
  const float* P3    = (const float*)d_in[12];
  const float* wgate = (const float*)d_in[13];
  const float* e_w1  = (const float*)d_in[14];
  const float* e_b1  = (const float*)d_in[15];
  const float* e_w2  = (const float*)d_in[16];
  const float* e_b2  = (const float*)d_in[17];

  char* ws = (char*)d_ws;
  size_t off = 0;
  auto alloc = [&](size_t bytes) -> void* {
    void* p = ws + off;
    off += (bytes + 255) & ~(size_t)255;
    return p;
  };

  float*  pooled = (float*) alloc(4608ull * 1024 * 4);
  __bf16* qbf    = (__bf16*)alloc(4608ull * 1024 * 2);
  __bf16* abf    = (__bf16*)alloc(4608ull * 4608 * 2);
  float*  dinv   = (float*) alloc(4608ull * 4);
  __bf16* xcpT   = (__bf16*)alloc(4608ull * 1024 * 2);   // [F,S] transposed
  __bf16* ybf    = (__bf16*)alloc(4608ull * 1024 * 2);   // [S,F]
  float*  X1     = (float*) alloc(4608ull * 1024 * 4);
  float*  X2     = (float*) alloc(1536ull * 512 * 4);
  float*  X3     = (float*) alloc(512ull * 256 * 4);
  __bf16* pt1    = (__bf16*)alloc(1024ull * 1024 * 2);   // P1^T [1024,1024]
  __bf16* pt2    = (__bf16*)alloc(512ull * 1024 * 2);    // P2^T [512,1024]
  __bf16* pt3    = (__bf16*)alloc(256ull * 512 * 2);     // P3^T [256,512]

  // projection matrices -> transposed bf16 (so all GEMMs are NT / K-major)
  f2bf_t_kernel<<<(1024 * 1024 + 255) / 256, 256, 0, stream>>>(P1, pt1, 1024, 1024);
  f2bf_t_kernel<<<(1024 * 512  + 255) / 256, 256, 0, stream>>>(P2, pt2, 1024, 512);
  f2bf_t_kernel<<<(512  * 256  + 255) / 256, 256, 0, stream>>>(P3, pt3, 512, 256);

  // ---------------- Stage 1: N=13824 -> S=4608, F=1024 ----------------
  attpool_kernel<<<4608, 256, 0, stream>>>(x, wp_w, wp_b, pooled, 1024);
  rownorm_kernel<<<4608, 256, 0, stream>>>(pooled, qbf, 1024);
  gemm_nt_bf16_kernel<1><<<dim3(36, 36), 256, 0, stream>>>(
      qbf, qbf, abf, nullptr, 4608, 4608, 1024, 1024, 1024, 4608);
  rowsum_dinv_kernel<<<4608, 256, 0, stream>>>(abf, dinv, 4608);
  nodeconv_t_kernel<<<(4608 * 1024 + 255) / 256, 256, 0, stream>>>(
      x, nc0_w, dinv, xcpT, 4608, 1024);
  gemm_nt_bf16_kernel<2><<<dim3(8, 36), 256, 0, stream>>>(
      abf, xcpT, ybf, dinv, 4608, 1024, 4608, 4608, 4608, 1024);
  gemm_nt_bf16_kernel<3><<<dim3(8, 36), 256, 0, stream>>>(
      ybf, pt1, X1, nullptr, 4608, 1024, 1024, 1024, 1024, 1024);

  // ---------------- Stage 2: S=1536, F=1024 -> 512 ----------------
  attpool_kernel<<<1536, 256, 0, stream>>>(X1, ap1_w, ap1_b, pooled, 1024);
  rownorm_kernel<<<1536, 256, 0, stream>>>(pooled, qbf, 1024);
  gemm_nt_bf16_kernel<1><<<dim3(12, 12), 256, 0, stream>>>(
      qbf, qbf, abf, nullptr, 1536, 1536, 1024, 1024, 1024, 1536);
  rowsum_dinv_kernel<<<1536, 256, 0, stream>>>(abf, dinv, 1536);
  nodeconv_t_kernel<<<(1536 * 1024 + 255) / 256, 256, 0, stream>>>(
      X1, nc1_w, dinv, xcpT, 1536, 1024);
  gemm_nt_bf16_kernel<2><<<dim3(8, 12), 256, 0, stream>>>(
      abf, xcpT, ybf, dinv, 1536, 1024, 1536, 1536, 1536, 1024);
  gemm_nt_bf16_kernel<3><<<dim3(4, 12), 256, 0, stream>>>(
      ybf, pt2, X2, nullptr, 1536, 512, 1024, 1024, 1024, 512);

  // ---------------- Stage 3: S=512, F=512 -> 256 ----------------
  attpool_kernel<<<512, 256, 0, stream>>>(X2, ap2_w, ap2_b, pooled, 512);
  rownorm_kernel<<<512, 256, 0, stream>>>(pooled, qbf, 512);
  gemm_nt_bf16_kernel<1><<<dim3(4, 4), 256, 0, stream>>>(
      qbf, qbf, abf, nullptr, 512, 512, 512, 512, 512, 512);
  rowsum_dinv_kernel<<<512, 256, 0, stream>>>(abf, dinv, 512);
  nodeconv_t_kernel<<<(512 * 512 + 255) / 256, 256, 0, stream>>>(
      X2, nc2_w, dinv, xcpT, 512, 512);
  gemm_nt_bf16_kernel<2><<<dim3(4, 4), 256, 0, stream>>>(
      abf, xcpT, ybf, dinv, 512, 512, 512, 512, 512, 512);
  gemm_nt_bf16_kernel<3><<<dim3(2, 4), 256, 0, stream>>>(
      ybf, pt3, X3, nullptr, 512, 256, 512, 512, 512, 256);

  // ---------------- MoE head: X3 viewed as [256, 512] ----------------
  moe_kernel<<<256, 64, 0, stream>>>(X3, wgate, e_w1, e_b1, e_w2, e_b2,
                                     (float*)d_out);
}
